// Mamba2_block_4140348473940
// MI455X (gfx1250) — compile-verified, matrix operand order checked by reading, not run on hardware
//
#include <hip/hip_runtime.h>
#include <hip/hip_bf16.h>

// ---------------------------------------------------------------------------
// Mamba2 block forward for MI455X (gfx1250, wave32, WMMA, async-to-LDS, TDM).
//
// Sizes (from reference): BATCH=B_*L_=8, S=2048, DM=1024, D_INNER=2048,
// D_STATE=128, NHEADS=32, HEADDIM=64, D_CONV=4, CHUNK=64, CONV_DIM=2304,
// D_IN_PROJ=4384.  Flat "rows" = 16384 = 8*2048 (u reshape is layout-preserving).
//
// Workspace requirement: ~690 MB (offsets computed below).
// ---------------------------------------------------------------------------

typedef __bf16 bf16_t;
typedef __attribute__((ext_vector_type(16))) __bf16 v16bf;
typedef __attribute__((ext_vector_type(8)))  __bf16 v8bf;
typedef __attribute__((ext_vector_type(8)))  float  v8f;
typedef __attribute__((ext_vector_type(4)))  unsigned int v4u;
typedef __attribute__((ext_vector_type(8)))  int v8i;
typedef __attribute__((ext_vector_type(4)))  int v4i;

#define ROWS    16384
#define DM      1024
#define DIP     4384
#define DI      2048
#define DST     128
#define NH      32
#define HD      64
#define CONVD   2304
#define SEQ     2048
#define CHUNK_L 64
#define NCHUNK  32

// ===========================================================================
// CDNA5 async copy helpers (ISA ch.10 async ops; tracked by ASYNCcnt).
// LDS destination operand: low 32 bits of a flat LDS pointer are the
// wave-relative LDS address (ISA 10.2 aperture mapping).
// ===========================================================================
__device__ __forceinline__ void async_ld_b128(const void* gsrc, void* lds_dst)
{
    asm volatile("global_load_async_to_lds_b128 %0, %1, off"
                 :: "v"((unsigned)(unsigned long long)lds_dst), "v"(gsrc)
                 : "memory");
}
__device__ __forceinline__ void wait_asynccnt0()
{
    asm volatile("s_wait_asynccnt 0x0" ::: "memory");
}

// ===========================================================================
// Tensor Data Mover: 2D tile load Global->LDS (ISA ch.7/8, VIMAGE encoding,
// tracked by TENSORcnt). D# built per ISA 8.3/8.4:
//   group0: [1:0]=count(1), [63:32]=lds_addr, [120:64]=global_addr, [127:126]=2
//   group1: [17:16]=data_size(1=2B), [79:48]=tensor_dim0, [111:80]=tensor_dim1,
//           [127:112]=tile_dim0, [143:128]=tile_dim1, [207:160]=dim0 stride
// OOB reads (rows >= tensor_dim1) return zero -> free N-edge masking.
// Tile is written to LDS contiguously: row stride = tile_dim0 elements.
// ===========================================================================
__device__ __forceinline__ void tdm_load_2d(unsigned lds_addr, const void* gaddr,
                                            unsigned tensor_d0, unsigned tensor_d1,
                                            unsigned tile_d0, unsigned tile_d1,
                                            unsigned stride0)
{
    unsigned long long ga = (unsigned long long)gaddr;
    v4u g0;
    g0[0] = 1u;                                               // count=1, user D#
    g0[1] = lds_addr;                                         // LDS byte address
    g0[2] = (unsigned)(ga & 0xffffffffu);                     // global_addr[31:0]
    g0[3] = (unsigned)((ga >> 32) & 0x01ffffffu) | (2u << 30);// addr[56:32]|type=2
    v8i g1;
    g1[0] = (int)(1u << 16);                                  // data_size = 2 bytes
    g1[1] = (int)((tensor_d0 & 0xffffu) << 16);               // tensor_dim0[15:0]
    g1[2] = (int)((tensor_d0 >> 16) | ((tensor_d1 & 0xffffu) << 16));
    g1[3] = (int)((tensor_d1 >> 16) | (tile_d0 << 16));       // tile_dim0
    g1[4] = (int)tile_d1;                                     // tile_dim1 (dim2=0)
    g1[5] = (int)stride0;                                     // dim0 stride lo
    g1[6] = 0;
    g1[7] = 0;
    v4i z4 = (v4i)0;
#if __clang_major__ >= 23
    v8i z8 = (v8i)0;
    __builtin_amdgcn_tensor_load_to_lds(g0, g1, z4, z4, z8, 0);
#else
    __builtin_amdgcn_tensor_load_to_lds(g0, g1, z4, z4, 0);
#endif
}
__device__ __forceinline__ void wait_tensorcnt0()
{
    __builtin_amdgcn_s_wait_tensorcnt((short)0);
}

// ===========================================================================
// 1) RMSNorm over D_MODEL + cast to bf16
// ===========================================================================
__global__ __launch_bounds__(256)
void k_rmsnorm_cast(const float* __restrict__ u, const float* __restrict__ w,
                    bf16_t* __restrict__ out)
{
    const int row = blockIdx.x;
    const float* x = u + (size_t)row * DM;
    __shared__ float red[256];
    float ss = 0.f;
    for (int i = threadIdx.x; i < DM; i += 256) { float v = x[i]; ss += v * v; }
    red[threadIdx.x] = ss;
    __syncthreads();
    for (int s = 128; s > 0; s >>= 1) {
        if (threadIdx.x < s) red[threadIdx.x] += red[threadIdx.x + s];
        __syncthreads();
    }
    const float scale = rsqrtf(red[0] / (float)DM + 1e-5f);
    for (int i = threadIdx.x; i < DM; i += 256)
        out[(size_t)row * DM + i] = (bf16_t)(x[i] * scale * w[i]);
}

// ===========================================================================
// 2) f32 -> bf16 cast
// ===========================================================================
__global__ __launch_bounds__(256)
void k_cast_bf16(const float* __restrict__ in, bf16_t* __restrict__ out, int n)
{
    int i = blockIdx.x * 256 + threadIdx.x;
    if (i < n) out[i] = (bf16_t)in[i];
}

// ===========================================================================
// 3) WMMA GEMM: C[M,N](f32) = A[M,K](bf16,row-major) * B[N,K](bf16,row-major)^T
//    Block tile 128x128, BK=64 (two WMMA K-steps per stage), 8 waves, each
//    wave 64x32 (4x2 tiles of 16x16) -> 16 v_wmma per stage.
//    Double-buffered LDS (64 KB):
//      - A tile: GLOBAL_LOAD_ASYNC_TO_LDS_B128 (all waves, ASYNCcnt)
//      - B tile: one TENSOR_LOAD_TO_LDS DMA issued by wave 0 (TENSORcnt);
//        TDM zero-fills rows beyond N (edge masking for free).
//    One s_wait_asynccnt / s_wait_tensorcnt + barrier per stage.
//    Optional residual add (resid indexed like C).
//    M,K multiples of 128/64 (true here); N edge handled by TDM + store guard.
// ===========================================================================
#define BM 128
#define BN 128
#define BK 64

__global__ __launch_bounds__(256)
void k_gemm_bf16(const bf16_t* __restrict__ A, const bf16_t* __restrict__ B,
                 float* __restrict__ C, const float* __restrict__ resid,
                 int M, int N, int K)
{
    __shared__ bf16_t As[2][BM * BK];   // 2 x 16 KB
    __shared__ bf16_t Bs[2][BN * BK];   // 2 x 16 KB

    const int t    = threadIdx.x;
    const int lane = t & 31;
    const int wave = t >> 5;
    const int wm   = (wave & 1) * 64;   // wave M offset within block tile
    const int wn   = (wave >> 1) * 32;  // wave N offset within block tile
    const int bm   = blockIdx.y * BM;
    const int bn   = blockIdx.x * BN;

    v8f acc[4][2];
    {
        v8f z;
        #pragma unroll
        for (int r = 0; r < 8; r++) z[r] = 0.f;
        #pragma unroll
        for (int i = 0; i < 4; i++)
            #pragma unroll
            for (int j = 0; j < 2; j++) acc[i][j] = z;
    }

    // A staging assignment: each thread moves 16 bf16 (2 x b128, at +0 and +16)
    const int lrow = t >> 1;          // 0..127 (tile row this thread stages)
    const int lcol = (t & 1) * 32;    // 0 or 32

    const int m0   = lane & 15;       // fragment row/col within 16
    const int half = lane >> 4;       // which K-half this lane holds

    auto stage = [&](int buf, int k0) {
        // A tile: per-lane async gather to LDS
        const bf16_t* sa = A + (size_t)(bm + lrow) * K + k0 + lcol;
        bf16_t* da = &As[buf][lrow * BK + lcol];
        async_ld_b128(sa,      da);
        async_ld_b128(sa + 16, da + 16);
        // B tile: single TDM descriptor, issued by wave 0 only
        if (wave == 0) {
            tdm_load_2d((unsigned)(unsigned long long)&Bs[buf][0],
                        B + (size_t)bn * K + k0,
                        (unsigned)K, (unsigned)N,   // tensor dims (elems, rows)
                        BK, BN,                     // tile dims
                        (unsigned)K);               // row stride (elems)
        }
    };

    stage(0, 0);
    int buf = 0;
    for (int k0 = 0; k0 < K; k0 += BK) {
        wait_asynccnt0();
        if (wave == 0) wait_tensorcnt0();
        __syncthreads();
        if (k0 + BK < K) stage(buf ^ 1, k0 + BK);

        #pragma unroll
        for (int kk = 0; kk < BK; kk += 32) {
            v16bf af[4], bfrag[2];
            #pragma unroll
            for (int i = 0; i < 4; i++) {
                const bf16_t* p = &As[buf][(wm + i * 16 + m0) * BK + kk];
                v8bf lo = *(const v8bf*)(p + half * 8);
                v8bf hi = *(const v8bf*)(p + 16 + half * 8);
                #pragma unroll
                for (int e = 0; e < 8; e++) { af[i][e] = lo[e]; af[i][e + 8] = hi[e]; }
            }
            #pragma unroll
            for (int j = 0; j < 2; j++) {
                const bf16_t* p = &Bs[buf][(wn + j * 16 + m0) * BK + kk];
                v8bf lo = *(const v8bf*)(p + half * 8);
                v8bf hi = *(const v8bf*)(p + 16 + half * 8);
                #pragma unroll
                for (int e = 0; e < 8; e++) { bfrag[j][e] = lo[e]; bfrag[j][e + 8] = hi[e]; }
            }
            #pragma unroll
            for (int i = 0; i < 4; i++)
                #pragma unroll
                for (int j = 0; j < 2; j++)
                    acc[i][j] = __builtin_amdgcn_wmma_f32_16x16x32_bf16(
                        false, af[i], false, bfrag[j], (short)0, acc[i][j],
                        false, false);
        }
        buf ^= 1;
    }

    // ---- epilogue: C/D layout: VGPR r -> row = i*16 + half*8 + r, col = m0 ----
    #pragma unroll
    for (int i = 0; i < 4; i++) {
        #pragma unroll
        for (int j = 0; j < 2; j++) {
            const int col = bn + wn + j * 16 + m0;
            if (col < N) {
                #pragma unroll
                for (int r = 0; r < 8; r++) {
                    const int row = bm + wm + i * 16 + half * 8 + r;
                    const size_t o = (size_t)row * N + col;
                    float v = acc[i][j][r];
                    if (resid) v += resid[o];
                    C[o] = v;
                }
            }
        }
    }
}

// ===========================================================================
// 4) dt = softplus(zx[:, 4352:4384] + dt_bias)  -> dtb[row*32+h]
// ===========================================================================
__global__ __launch_bounds__(256)
void k_dt(const float* __restrict__ zx, const float* __restrict__ dt_bias,
          float* __restrict__ dtb)
{
    int i = blockIdx.x * 256 + threadIdx.x;
    if (i < ROWS * NH) {
        int row = i >> 5, h = i & 31;
        float v = zx[(size_t)row * DIP + (DI + CONVD) + h] + dt_bias[h];
        dtb[i] = (v > 20.f) ? v : log1pf(expf(v));
    }
}

// ===========================================================================
// 5) depthwise causal conv (width 4) over seq + SiLU.
//    xBC lives inside zx at column offset D_INNER; output xc[row][c].
// ===========================================================================
__global__ __launch_bounds__(256)
void k_conv(const float* __restrict__ zx, const float* __restrict__ cw,
            const float* __restrict__ cb, float* __restrict__ xc)
{
    const int c = blockIdx.x * 256 + threadIdx.x;   // channel 0..2303
    const int row = blockIdx.y;                     // 0..16383
    if (c >= CONVD) return;
    const int b2 = row >> 11;                       // row / 2048
    const int s  = row & 2047;
    float acc = cb[c];
    #pragma unroll
    for (int j = 0; j < 4; j++) {
        int si = s - 3 + j;
        if (si >= 0)
            acc += zx[(size_t)(b2 * SEQ + si) * DIP + DI + c] * cw[c * 4 + j];
    }
    const float sig = 1.f / (1.f + expf(-acc));
    xc[(size_t)row * CONVD + c] = acc * sig;
}

// ===========================================================================
// 6) SSD chunked scan. One workgroup per (head, batch); sequential over 32
//    chunks. State [HD=64][DST=128] f32 lives in LDS; per-chunk A_cum via a
//    wave32 __shfl inclusive scan (redundantly per wave, no LDS).
//    LDS total = 32K (S) + 16K (B bf16) + 8K (xdt bf16) + 8K (M bf16) = 64 KB.
// ===========================================================================
__global__ __launch_bounds__(256)
void k_ssd(const float* __restrict__ xc, const float* __restrict__ dtb,
           const float* __restrict__ A_log, const float* __restrict__ Dp,
           float* __restrict__ y)
{
    __shared__ float  S [HD * DST];      // state  [p][n]
    __shared__ bf16_t sB[CHUNK_L * DST]; // B      [l][n]
    __shared__ bf16_t sX[CHUNK_L * HD];  // x*dt   [l][p]
    __shared__ bf16_t sM[CHUNK_L * CHUNK_L]; // masked G*L [l][s]

    const int h    = blockIdx.x;         // head  0..31
    const int b2   = blockIdx.y;         // batch 0..7
    const int t    = threadIdx.x;
    const int lane = t & 31;
    const float Ah = -expf(A_log[h]);
    const float Dh = Dp[h];

    for (int i = t; i < HD * DST; i += 256) S[i] = 0.f;
    __syncthreads();

    const int lfix = t >> 2;             // fixed l (or p) per thread, 0..63
    const int q    = t & 3;              // quarter

    for (int c = 0; c < NCHUNK; ++c) {
        const int rowbase = b2 * SEQ + c * CHUNK_L;

        // ---- per-wave inclusive scan of a_l = Ah*dt_l over l=0..63 ----
        float dt0 = dtb[(size_t)(rowbase + lane) * NH + h];
        float dt1 = dtb[(size_t)(rowbase + 32 + lane) * NH + h];
        float s0 = Ah * dt0;
        #pragma unroll
        for (int off = 1; off < 32; off <<= 1) {
            float tmp = __shfl_up(s0, off, 32);
            if (lane >= off) s0 += tmp;
        }
        const float tot0 = __shfl(s0, 31, 32);
        float s1 = Ah * dt1;
        #pragma unroll
        for (int off = 1; off < 32; off <<= 1) {
            float tmp = __shfl_up(s1, off, 32);
            if (lane >= off) s1 += tmp;
        }
        s1 += tot0;
        const float Asum = __shfl(s1, 31, 32);

        auto acum = [&](int l) -> float {
            float lo = __shfl(s0, l & 31, 32);
            float hi = __shfl(s1, l & 31, 32);
            return (l < 32) ? lo : hi;
        };

        // ---- stage B tile and xdt tile into LDS (bf16) ----
        {
            const int n0 = q * 32;
            const float* src = xc + (size_t)(rowbase + lfix) * CONVD + DI + n0;
            #pragma unroll 8
            for (int i = 0; i < 32; i++)
                sB[lfix * DST + n0 + i] = (bf16_t)src[i];
        }
        {
            float dl_lo = __shfl(dt0, lfix & 31, 32);
            float dl_hi = __shfl(dt1, lfix & 31, 32);
            const float dtl = (lfix < 32) ? dl_lo : dl_hi;
            const float* src = xc + (size_t)(rowbase + lfix) * CONVD + h * HD + q * 16;
            #pragma unroll
            for (int i = 0; i < 16; i++)
                sX[lfix * HD + q * 16 + i] = (bf16_t)(src[i] * dtl);
        }
        __syncthreads();

        // ---- Y_off (uses previous state): Y[l,p] = expA(l) * sum_n C[l,n]*S[p,n]
        float Yacc[16];
        const float acl   = acum(lfix);
        const float expAl = expf(acl);
        const float* Crow = xc + (size_t)(rowbase + lfix) * CONVD + DI + DST;
        {
            #pragma unroll
            for (int i = 0; i < 16; i++) Yacc[i] = 0.f;
            for (int n = 0; n < DST; ++n) {
                const float cv = Crow[n];
                #pragma unroll
                for (int i = 0; i < 16; i++)
                    Yacc[i] += cv * S[(q * 16 + i) * DST + n];
            }
            #pragma unroll
            for (int i = 0; i < 16; i++) Yacc[i] *= expAl;
        }

        // ---- M[l,s] = (s<=l) ? (C_l . B_s) * exp(A_cum[l]-A_cum[s]) : 0 ----
        {
            float g[16];
            #pragma unroll
            for (int i = 0; i < 16; i++) g[i] = 0.f;
            for (int n = 0; n < DST; ++n) {
                const float cv = Crow[n];
                #pragma unroll
                for (int i = 0; i < 16; i++)
                    g[i] += cv * (float)sB[(q * 16 + i) * DST + n];
            }
            #pragma unroll
            for (int i = 0; i < 16; i++) {
                const int sidx = q * 16 + i;
                float m = (sidx <= lfix) ? g[i] * expf(acl - acum(sidx)) : 0.f;
                sM[lfix * CHUNK_L + sidx] = (bf16_t)m;
            }
        }
        __syncthreads();

        // ---- Y_diag: Y[l,p] += sum_s M[l,s]*xdt[s,p]; write y out ----
        {
            for (int sidx = 0; sidx <= lfix; ++sidx) {
                const float m = (float)sM[lfix * CHUNK_L + sidx];
                #pragma unroll
                for (int i = 0; i < 16; i++)
                    Yacc[i] += m * (float)sX[sidx * HD + q * 16 + i];
            }
            const float* xrow = xc + (size_t)(rowbase + lfix) * CONVD + h * HD + q * 16;
            float* yrow = y + (size_t)(rowbase + lfix) * DI + h * HD + q * 16;
            #pragma unroll
            for (int i = 0; i < 16; i++)
                yrow[i] = Yacc[i] + xrow[i] * Dh;
        }

        // ---- state update: S[p,n] = exp(Asum)*S + sum_l B[l,n]*exp(Asum-A_cum[l])*xdt[l,p]
        {
            const int p = lfix, n0 = q * 32;
            const float dAsum = expf(Asum);
            float sreg[32];
            #pragma unroll 8
            for (int i = 0; i < 32; i++) sreg[i] = S[p * DST + n0 + i] * dAsum;
            for (int l = 0; l < CHUNK_L; ++l) {
                const float coef = expf(Asum - acum(l)) * (float)sX[l * HD + p];
                #pragma unroll 8
                for (int i = 0; i < 32; i++)
                    sreg[i] += coef * (float)sB[l * DST + n0 + i];
            }
            #pragma unroll 8
            for (int i = 0; i < 32; i++) S[p * DST + n0 + i] = sreg[i];
        }
        __syncthreads();
    }
}

// ===========================================================================
// 7) y_gated = y * silu(z); RMSNorm over D_INNER with norm_w; cast bf16
// ===========================================================================
__global__ __launch_bounds__(256)
void k_gate_rms(const float* __restrict__ y, const float* __restrict__ zx,
                const float* __restrict__ nw, bf16_t* __restrict__ yn)
{
    const int row = blockIdx.x;
    __shared__ float red[256];
    float ss = 0.f;
    for (int i = threadIdx.x; i < DI; i += 256) {
        const float z = zx[(size_t)row * DIP + i];
        const float g = z / (1.f + expf(-z));
        const float v = y[(size_t)row * DI + i] * g;
        ss += v * v;
    }
    red[threadIdx.x] = ss;
    __syncthreads();
    for (int s = 128; s > 0; s >>= 1) {
        if (threadIdx.x < s) red[threadIdx.x] += red[threadIdx.x + s];
        __syncthreads();
    }
    const float scale = rsqrtf(red[0] / (float)DI + 1e-5f);
    for (int i = threadIdx.x; i < DI; i += 256) {
        const float z = zx[(size_t)row * DIP + i];
        const float g = z / (1.f + expf(-z));
        const float v = y[(size_t)row * DI + i] * g;
        yn[(size_t)row * DI + i] = (bf16_t)(v * scale * nw[i]);
    }
}

// ===========================================================================
// launcher
// ===========================================================================
extern "C" void kernel_launch(void* const* d_in, const int* in_sizes, int n_in,
                              void* d_out, int out_size, void* d_ws, size_t ws_size,
                              hipStream_t stream)
{
    const float* u          = (const float*)d_in[0];
    const float* in_proj_w  = (const float*)d_in[1];
    const float* conv_w     = (const float*)d_in[2];
    const float* conv_b     = (const float*)d_in[3];
    const float* pre_norm_w = (const float*)d_in[4];
    const float* dt_bias    = (const float*)d_in[5];
    const float* A_log      = (const float*)d_in[6];
    const float* Dvec       = (const float*)d_in[7];
    const float* norm_w     = (const float*)d_in[8];
    const float* out_proj_w = (const float*)d_in[9];
    float* out = (float*)d_out;

    // ---- workspace layout (~690 MB) ----
    char* ws = (char*)d_ws;
    size_t off = 0;
    auto alloc = [&](size_t bytes) -> char* {
        char* p = ws + off;
        off = (off + bytes + 255) & ~(size_t)255;
        return p;
    };
    bf16_t* un    = (bf16_t*)alloc((size_t)ROWS * DM * 2);
    bf16_t* Winb  = (bf16_t*)alloc((size_t)DIP * DM * 2);
    bf16_t* Woutb = (bf16_t*)alloc((size_t)DM * DI * 2);
    float*  zx    = (float*) alloc((size_t)ROWS * DIP * 4);
    float*  dtb   = (float*) alloc((size_t)ROWS * NH * 4);
    float*  xc    = (float*) alloc((size_t)ROWS * CONVD * 4);
    float*  ybuf  = (float*) alloc((size_t)ROWS * DI * 4);
    bf16_t* ynb   = (bf16_t*)alloc((size_t)ROWS * DI * 2);
    (void)ws_size; (void)in_sizes; (void)n_in; (void)out_size;

    // 1) pre-RMSNorm + bf16 cast of activations
    k_rmsnorm_cast<<<ROWS, 256, 0, stream>>>(u, pre_norm_w, un);

    // 2) weight casts
    k_cast_bf16<<<((DIP * DM) + 255) / 256, 256, 0, stream>>>(in_proj_w, Winb, DIP * DM);
    k_cast_bf16<<<((DM * DI) + 255) / 256, 256, 0, stream>>>(out_proj_w, Woutb, DM * DI);

    // 3) in_proj GEMM: zx[16384,4384] = un[16384,1024] x Win^T
    {
        dim3 grid((DIP + BN - 1) / BN, ROWS / BM);
        k_gemm_bf16<<<grid, 256, 0, stream>>>(un, Winb, zx, nullptr, ROWS, DIP, DM);
    }

    // 4) dt = softplus(...)
    k_dt<<<(ROWS * NH + 255) / 256, 256, 0, stream>>>(zx, dt_bias, dtb);

    // 5) depthwise conv + SiLU
    {
        dim3 grid(CONVD / 256, ROWS);
        k_conv<<<grid, 256, 0, stream>>>(zx, conv_w, conv_b, xc);
    }

    // 6) SSD chunked scan
    {
        dim3 grid(NH, ROWS / SEQ);
        k_ssd<<<grid, 256, 0, stream>>>(xc, dtb, A_log, Dvec, ybuf);
    }

    // 7) gate + RMSNorm + bf16 cast
    k_gate_rms<<<ROWS, 256, 0, stream>>>(ybuf, zx, norm_w, ynb);

    // 8) out_proj GEMM + residual: out = u + yn x Wout^T
    {
        dim3 grid(DM / BN, ROWS / BM);
        k_gemm_bf16<<<grid, 256, 0, stream>>>(ynb, Woutb, out, u, ROWS, DM, DI);
    }
}
